// Block_5480378270028
// MI455X (gfx1250) — matrix operand order repaired and checked by reading, not compiled, not run on hardware
//
#include <hip/hip_runtime.h>
#include <hip/hip_bf16.h>

// ---------------------------------------------------------------------------
// Transformer block (pre-LN attn + MLP) for MI455X (gfx1250, wave32, WMMA)
// B=4, N=2048, C=768, H=12, DH=64, HID=3072.  All GEMMs via
// v_wmma_f32_16x16x32_bf16 (fp32 accum); weight tiles staged in LDS via
// global_load_async_to_lds_b128 (ASYNCcnt) with double buffering.
// ---------------------------------------------------------------------------

typedef __bf16 bf16_t;
typedef __attribute__((ext_vector_type(16))) __bf16 v16bf;
typedef __attribute__((ext_vector_type(8)))  __bf16 v8bf;
typedef __attribute__((ext_vector_type(8)))  float  v8f;

#define NB   2048   // tokens per batch
#define CB   768    // channels
#define HB   12     // heads
#define DHB  64     // head dim
#define HIDB 3072   // mlp hidden
#define MB   8192   // B*N rows

__device__ __forceinline__ v8f wmma_bf16(v16bf a, v16bf b, v8f c) {
  // D = A(16x32 bf16) * B(32x16 bf16) + C(16x16 f32)
  return __builtin_amdgcn_wmma_f32_16x16x32_bf16(
      /*neg_a=*/false, a, /*neg_b=*/false, b,
      /*c_mod=*/(short)0, c, /*reuse_a=*/false, /*reuse_b=*/false);
}

// A-matrix 16x32 bf16 fragment (ISA 7.12.2):
// lanes 0-15 : row M = caller row, K = {k0..k0+7, k0+16..k0+23}
// lanes 16-31: same M,             K shifted +8
__device__ __forceinline__ v16bf load_fragA(const bf16_t* __restrict__ base,
                                            int row, int ld, int k0, int lane) {
  const bf16_t* p = base + (size_t)row * (size_t)ld + (size_t)(k0 + ((lane >> 4) << 3));
  v8bf lo = *(const v8bf*)p;
  v8bf hi = *(const v8bf*)(p + 16);
  v16bf r;
#pragma unroll
  for (int i = 0; i < 8; ++i) { r[i] = lo[i]; r[8 + i] = hi[i]; }
  return r;
}

// B-matrix 32x16 bf16 fragment: per lane N = col (caller row on B^T storage),
// lanes 0-15 hold K = {k0..k0+7, k0+16..k0+23}, lanes 16-31 same K shifted +8.
__device__ __forceinline__ v16bf load_fragB(const bf16_t* __restrict__ base,
                                            int row, int ld, int k0, int lane) {
  const bf16_t* p = base + (size_t)row * (size_t)ld + (size_t)(k0 + ((lane >> 4) << 3));
  v8bf lo = *(const v8bf*)p;
  v8bf hi = *(const v8bf*)(p + 16);
  v16bf r;
#pragma unroll
  for (int i = 0; i < 8; ++i) { r[i] = lo[i]; r[8 + i] = hi[i]; }
  return r;
}

// ---------------------------------------------------------------------------
// Weight transpose + fp32 -> bf16 convert:  w[K,N] -> wt[N,K]
// ---------------------------------------------------------------------------
__global__ void convT_kernel(const float* __restrict__ w, bf16_t* __restrict__ wt,
                             int K, int Nn) {
  int idx = blockIdx.x * blockDim.x + threadIdx.x;
  int n = idx / K;
  int k = idx - n * K;
  wt[(size_t)n * K + k] = (bf16_t)w[(size_t)k * Nn + n];
}

// ---------------------------------------------------------------------------
// LayerNorm over C=768, fp32 in -> bf16 out.  One 256-thread block per row.
// ---------------------------------------------------------------------------
__global__ __launch_bounds__(256) void ln_kernel(const float* __restrict__ X,
                                                 const float* __restrict__ g,
                                                 const float* __restrict__ bta,
                                                 bf16_t* __restrict__ out) {
  __shared__ float s1[256], s2[256];
  const int row = blockIdx.x, tid = threadIdx.x;
  const float* xr = X + (size_t)row * CB;
  float v[3], a = 0.f, b = 0.f;
#pragma unroll
  for (int j = 0; j < 3; ++j) { v[j] = xr[tid + j * 256]; a += v[j]; b += v[j] * v[j]; }
  s1[tid] = a; s2[tid] = b;
  __syncthreads();
  for (int st = 128; st > 0; st >>= 1) {
    if (tid < st) { s1[tid] += s1[tid + st]; s2[tid] += s2[tid + st]; }
    __syncthreads();
  }
  const float mu = s1[0] * (1.0f / CB);
  const float var = s2[0] * (1.0f / CB) - mu * mu;
  const float rstd = rsqrtf(var + 1e-5f);
  bf16_t* orow = out + (size_t)row * CB;
#pragma unroll
  for (int j = 0; j < 3; ++j) {
    int i = tid + j * 256;
    orow[i] = (bf16_t)((v[j] - mu) * rstd * g[i] + bta[i]);
  }
}

// ---------------------------------------------------------------------------
// LDS-staged WMMA GEMM: out[M,Nn] = A[M,K](bf16) * BT[Nn,K](bf16)^T
// 256 threads = 8 waves; block tile 256x64 (wave tile 32x64), K step 32.
// The shared 64x32 B tile is streamed into LDS once per block per K-step with
// global_load_async_to_lds_b128 (one async b128 per wave = 512B; 8 waves fill
// the 4KB tile), double-buffered: s_wait_asynccnt 1 after issuing the next
// buffer guarantees the current buffer is resident before the barrier.
// EPI 0: fp32 + bias + residual.  EPI 1: bias + exact GELU -> bf16.
// EPI 2: QKV head-split scatter -> q(scaled)/k [B,H,N,DH], vT [B,H,DH,N].
// ---------------------------------------------------------------------------
template <int EPI>
__global__ __launch_bounds__(256) void gemm_lds(const bf16_t* __restrict__ A,
                                                const bf16_t* __restrict__ BT,
                                                const float* __restrict__ bias,
                                                const float* __restrict__ resid,
                                                float* __restrict__ outF,
                                                bf16_t* __restrict__ outB,
                                                bf16_t* __restrict__ qo,
                                                bf16_t* __restrict__ ko,
                                                bf16_t* __restrict__ vto,
                                                int Nn, int K) {
  __shared__ __align__(16) bf16_t Bs[2][64 * 32];
  const int tid = threadIdx.x;
  const int lane = tid & 31, wave = tid >> 5;
  const int m0 = blockIdx.y * 256 + wave * 32;
  const int n0 = blockIdx.x * 64;

  // Async-copy mapping: thread tid moves 16B; row = tid/4, chunk = (tid%4)*8.
  const int brow = tid >> 2;
  const int bchunk = (tid & 3) << 3;
  const unsigned gbase_off = (unsigned)(((n0 + brow) * K + bchunk) * 2);

  auto issue = [&](int buf, int k0) {
    unsigned ldsa = (unsigned)(size_t)&Bs[buf][brow * 32 + bchunk];
    unsigned voff = gbase_off + (unsigned)(k0 * 2);
    asm volatile("global_load_async_to_lds_b128 %0, %1, %2"
                 :: "v"(ldsa), "v"(voff), "s"(BT)
                 : "memory");
  };

  v8f acc[2][4] = {};
  const int nsteps = K / 32;
  issue(0, 0);
  for (int s = 0; s < nsteps; ++s) {
    const int k0 = s * 32;
    if (s + 1 < nsteps) {
      issue((s + 1) & 1, k0 + 32);
      asm volatile("s_wait_asynccnt 1" ::: "memory");
    } else {
      asm volatile("s_wait_asynccnt 0" ::: "memory");
    }
    __syncthreads();                       // current B tile visible block-wide
    const bf16_t* bs = Bs[s & 1];
    v16bf a0 = load_fragA(A, m0 + (lane & 15), K, k0, lane);
    v16bf a1 = load_fragA(A, m0 + 16 + (lane & 15), K, k0, lane);
#pragma unroll
    for (int j = 0; j < 4; ++j) {
      v16bf b = load_fragB(bs, j * 16 + (lane & 15), 32, 0, lane);
      acc[0][j] = wmma_bf16(a0, b, acc[0][j]);
      acc[1][j] = wmma_bf16(a1, b, acc[1][j]);
    }
    __syncthreads();                       // done reading before next overwrite
  }

#pragma unroll
  for (int rb = 0; rb < 2; ++rb) {
#pragma unroll
    for (int j = 0; j < 4; ++j) {
#pragma unroll
      for (int i = 0; i < 8; ++i) {
        const int r = m0 + rb * 16 + i + ((lane >> 4) << 3);
        const int c = n0 + j * 16 + (lane & 15);
        const float raw = acc[rb][j][i];
        if constexpr (EPI == 0) {
          float v = raw + (bias ? bias[c] : 0.0f);
          outF[(size_t)r * Nn + c] = v + (resid ? resid[(size_t)r * Nn + c] : 0.0f);
        } else if constexpr (EPI == 1) {
          float v = raw + (bias ? bias[c] : 0.0f);
          outB[(size_t)r * Nn + c] = (bf16_t)(0.5f * v * (1.0f + erff(v * 0.70710678118f)));
        } else {
          const int which = c / CB, cc = c - which * CB;
          const int head = cc >> 6, d = cc & 63;
          const int bb = r >> 11, t = r & (NB - 1);
          const size_t bh = (size_t)bb * HB + head;
          if (which == 0)      qo[(bh * NB + t) * DHB + d] = (bf16_t)(raw * 0.125f); // DH^-0.5
          else if (which == 1) ko[(bh * NB + t) * DHB + d] = (bf16_t)raw;
          else                 vto[(bh * DHB + d) * NB + t] = (bf16_t)raw;
        }
      }
    }
  }
}

// ---------------------------------------------------------------------------
// Flash attention: one wave owns 16 query rows of one (b,h).
// Per 32-KV step: 4 WMMA for S = Q*K^T, online softmax (shfl row reductions),
// P staged through per-wave 16x32 LDS tile to form A-fragment, 4 WMMA for P*V.
// ---------------------------------------------------------------------------
__global__ __launch_bounds__(128) void attn_kernel(const bf16_t* __restrict__ Q,
                                                   const bf16_t* __restrict__ Kb,
                                                   const bf16_t* __restrict__ VT,
                                                   bf16_t* __restrict__ Y) {
  __shared__ __align__(16) bf16_t plds[4][16 * 32];
  const int lane = threadIdx.x & 31;
  const int wave = threadIdx.x >> 5;
  const int tile = blockIdx.x * 4 + wave;        // 0 .. B*H*(N/16)-1
  const int bh = tile >> 7;                      // tile / (N/16)
  const int t0 = (tile & 127) << 4;              // query-row block start
  const bf16_t* q  = Q  + (size_t)bh * NB * DHB;
  const bf16_t* k  = Kb + (size_t)bh * NB * DHB;
  const bf16_t* vt = VT + (size_t)bh * DHB * NB;
  bf16_t* pl = plds[wave];

  const v16bf qa0 = load_fragA(q, t0 + (lane & 15), DHB, 0,  lane);
  const v16bf qa1 = load_fragA(q, t0 + (lane & 15), DHB, 32, lane);

  v8f o0 = {}, o1 = {}, o2 = {}, o3 = {};
  float m[8], l[8];
#pragma unroll
  for (int i = 0; i < 8; ++i) { m[i] = -3.0e38f; l[i] = 0.f; }

  for (int kv0 = 0; kv0 < NB; kv0 += 32) {
    v8f s0 = {}, s1 = {};
    {
      v16bf kb = load_fragB(k, kv0 + (lane & 15), DHB, 0, lane);
      s0 = wmma_bf16(qa0, kb, s0);
      kb = load_fragB(k, kv0 + (lane & 15), DHB, 32, lane);
      s0 = wmma_bf16(qa1, kb, s0);
      kb = load_fragB(k, kv0 + 16 + (lane & 15), DHB, 0, lane);
      s1 = wmma_bf16(qa0, kb, s1);
      kb = load_fragB(k, kv0 + 16 + (lane & 15), DHB, 32, lane);
      s1 = wmma_bf16(qa1, kb, s1);
    }
#pragma unroll
    for (int i = 0; i < 8; ++i) {
      float tmax = fmaxf(s0[i], s1[i]);
#pragma unroll
      for (int off = 1; off < 16; off <<= 1)
        tmax = fmaxf(tmax, __shfl_xor(tmax, off, 32));
      const float mnew = fmaxf(m[i], tmax);
      const float p0 = __expf(s0[i] - mnew);
      const float p1 = __expf(s1[i] - mnew);
      const float sc = __expf(m[i] - mnew);
      float rs = p0 + p1;
#pragma unroll
      for (int off = 1; off < 16; off <<= 1)
        rs += __shfl_xor(rs, off, 32);
      l[i] = l[i] * sc + rs;
      m[i] = mnew;
      const int r = i + ((lane >> 4) << 3);
      pl[r * 32 + (lane & 15)]      = (bf16_t)p0;
      pl[r * 32 + 16 + (lane & 15)] = (bf16_t)p1;
      o0[i] *= sc; o1[i] *= sc; o2[i] *= sc; o3[i] *= sc;
    }
    asm volatile("s_wait_dscnt 0" ::: "memory");   // wave-wide P stores visible
    const v16bf pa = load_fragA(pl, (lane & 15), 32, 0, lane);
    v16bf vb = load_fragB(vt, 0 + (lane & 15), NB, kv0, lane);
    o0 = wmma_bf16(pa, vb, o0);
    vb = load_fragB(vt, 16 + (lane & 15), NB, kv0, lane);
    o1 = wmma_bf16(pa, vb, o1);
    vb = load_fragB(vt, 32 + (lane & 15), NB, kv0, lane);
    o2 = wmma_bf16(pa, vb, o2);
    vb = load_fragB(vt, 48 + (lane & 15), NB, kv0, lane);
    o3 = wmma_bf16(pa, vb, o3);
  }

  const int batch = bh / HB, head = bh - batch * HB;
#pragma unroll
  for (int i = 0; i < 8; ++i) {
    const float inv = 1.0f / l[i];
    const int token = t0 + i + ((lane >> 4) << 3);
    const size_t o = ((size_t)batch * NB + token) * CB + head * DHB + (lane & 15);
    Y[o + 0]  = (bf16_t)(o0[i] * inv);
    Y[o + 16] = (bf16_t)(o1[i] * inv);
    Y[o + 32] = (bf16_t)(o2[i] * inv);
    Y[o + 48] = (bf16_t)(o3[i] * inv);
  }
}

// ---------------------------------------------------------------------------
extern "C" void kernel_launch(void* const* d_in, const int* in_sizes, int n_in,
                              void* d_out, int out_size, void* d_ws, size_t ws_size,
                              hipStream_t stream) {
  (void)in_sizes; (void)n_in; (void)out_size; (void)ws_size;
  const float* x      = (const float*)d_in[0];
  const float* ln1_g  = (const float*)d_in[1];
  const float* ln1_b  = (const float*)d_in[2];
  const float* w_qkv  = (const float*)d_in[3];
  const float* w_proj = (const float*)d_in[4];
  const float* b_proj = (const float*)d_in[5];
  const float* ln2_g  = (const float*)d_in[6];
  const float* ln2_b  = (const float*)d_in[7];
  const float* w_fc1  = (const float*)d_in[8];
  const float* b_fc1  = (const float*)d_in[9];
  const float* w_fc2  = (const float*)d_in[10];
  const float* b_fc2  = (const float*)d_in[11];
  float* out = (float*)d_out;

  char* ws = (char*)d_ws;
  size_t off = 0;
  auto take = [&](size_t elems, size_t esz) {
    void* p = ws + off;
    off += ((elems * esz) + 255) & ~(size_t)255;
    return p;
  };
  bf16_t* wqkvT = (bf16_t*)take((size_t)3 * CB * CB, 2);   // [2304,768]
  bf16_t* wprjT = (bf16_t*)take((size_t)CB * CB, 2);       // [768,768]
  bf16_t* wfc1T = (bf16_t*)take((size_t)HIDB * CB, 2);     // [3072,768]
  bf16_t* wfc2T = (bf16_t*)take((size_t)CB * HIDB, 2);     // [768,3072]
  bf16_t* h1    = (bf16_t*)take((size_t)MB * CB, 2);
  bf16_t* qb    = (bf16_t*)take((size_t)MB * CB, 2);       // [B,H,N,DH]
  bf16_t* kb    = (bf16_t*)take((size_t)MB * CB, 2);
  bf16_t* vtb   = (bf16_t*)take((size_t)MB * CB, 2);       // [B,H,DH,N]
  bf16_t* yatt  = (bf16_t*)take((size_t)MB * CB, 2);
  float*  x2    = (float*) take((size_t)MB * CB, 4);
  bf16_t* h2    = (bf16_t*)take((size_t)MB * CB, 2);
  bf16_t* hmid  = (bf16_t*)take((size_t)MB * HIDB, 2);

  // 1) weight transpose/convert to bf16 [out,in]
  convT_kernel<<<(CB * 3 * CB) / 256, 256, 0, stream>>>(w_qkv, wqkvT, CB, 3 * CB);
  convT_kernel<<<(CB * CB) / 256, 256, 0, stream>>>(w_proj, wprjT, CB, CB);
  convT_kernel<<<(CB * HIDB) / 256, 256, 0, stream>>>(w_fc1, wfc1T, CB, HIDB);
  convT_kernel<<<(HIDB * CB) / 256, 256, 0, stream>>>(w_fc2, wfc2T, HIDB, CB);

  // 2) LN1
  ln_kernel<<<MB, 256, 0, stream>>>(x, ln1_g, ln1_b, h1);

  // 3) QKV GEMM + head scatter (q pre-scaled)
  gemm_lds<2><<<dim3((3 * CB) / 64, MB / 256), 256, 0, stream>>>(
      h1, wqkvT, nullptr, nullptr, nullptr, nullptr, qb, kb, vtb, 3 * CB, CB);

  // 4) flash attention -> yatt bf16 [M,C]
  attn_kernel<<<1536, 128, 0, stream>>>(qb, kb, vtb, yatt);

  // 5) proj + bias + residual(x) -> x2 fp32
  gemm_lds<0><<<dim3(CB / 64, MB / 256), 256, 0, stream>>>(
      yatt, wprjT, b_proj, x, x2, nullptr, nullptr, nullptr, nullptr, CB, CB);

  // 6) LN2
  ln_kernel<<<MB, 256, 0, stream>>>(x2, ln2_g, ln2_b, h2);

  // 7) FC1 + bias + exact GELU -> hmid bf16
  gemm_lds<1><<<dim3(HIDB / 64, MB / 256), 256, 0, stream>>>(
      h2, wfc1T, b_fc1, nullptr, nullptr, hmid, nullptr, nullptr, nullptr, HIDB, CB);

  // 8) FC2 + bias + residual(x2) -> out fp32
  gemm_lds<0><<<dim3(CB / 64, MB / 256), 256, 0, stream>>>(
      hmid, wfc2T, b_fc2, x2, out, nullptr, nullptr, nullptr, nullptr, CB, HIDB);
}
